// KnnSfmxConstLoss_30210799960502
// MI455X (gfx1250) — matrix operand order repaired and checked by reading, not compiled, not run on hardware
//
#include <hip/hip_runtime.h>
#include <hip/hip_bf16.h>

#define N_SRC 4096
#define N_TAR 4096
#define DIM   1024
#define NUM_CLS 64
#define TOPS  5
#define RK    10
#define TOPN  2048
#define NEG_FILL (-2.0f)
#define BROW_STRIDE 40   // 32 K-elems + 8 pad elems (80B) -> conflict-free ds_load_b128

typedef __attribute__((ext_vector_type(8)))  __bf16 v8bf;
typedef __attribute__((ext_vector_type(16))) __bf16 v16bf;
typedef __attribute__((ext_vector_type(8)))  float  v8f;

union FragBF { v16bf v; v8bf h[2]; };

// ---------------- K1: normalize rows + bf16 hi/lo split -------------------
__global__ __launch_bounds__(256)
void normalize_split_kernel(const float* __restrict__ x,
                            __bf16* __restrict__ hi, __bf16* __restrict__ lo) {
    const int row = blockIdx.x;
    const int t = threadIdx.x;
    const float* xr = x + (size_t)row * DIM;
    float vals[4];
    float ss = 0.0f;
    for (int p = 0; p < 4; ++p) {
        vals[p] = xr[p * 256 + t];
        ss += vals[p] * vals[p];
    }
    __shared__ float red[256];
    red[t] = ss;
    __syncthreads();
    for (int s = 128; s >= 1; s >>= 1) {
        if (t < s) red[t] += red[t + s];
        __syncthreads();
    }
    const float inv = 1.0f / sqrtf(red[0]);
    for (int p = 0; p < 4; ++p) {
        float v = vals[p] * inv;
        __bf16 h = (__bf16)v;
        __bf16 l = (__bf16)(v - (float)h);
        size_t idx = (size_t)row * DIM + p * 256 + t;
        hi[idx] = h;
        lo[idx] = l;
    }
}

// ---------------- K2: simT = 0.5*(T_norm @ S_norm^T + 1) via bf16x3 WMMA ----
// A fragments: direct global loads (no cross-wave reuse).
// B tile: async-copied global->LDS, double buffered, shared by all 8 waves.
__device__ inline v16bf load_fragA(const __bf16* base, int row, int k0, int half) {
    // 16-bit A layout: lanes 0-15: K = {k0..k0+7, k0+16..k0+23}; lanes 16-31: +8
    const __bf16* p = base + (size_t)row * DIM + k0 + 8 * half;
    FragBF f;
    f.h[0] = *(const v8bf*)(p);
    f.h[1] = *(const v8bf*)(p + 16);
    return f.v;
}
__device__ inline v16bf load_fragB_lds(const __bf16* tile, int nloc, int half) {
    // 16-bit B layout: lanes 0-15: K = {0..7, 16..23}; lanes 16-31: +8
    const __bf16* p = tile + nloc * BROW_STRIDE + 8 * half;
    FragBF f;
    f.h[0] = *(const v8bf*)(p);
    f.h[1] = *(const v8bf*)(p + 16);
    return f.v;
}
__device__ inline v8f wmma_bf16(v16bf a, v16bf b, v8f c) {
    return __builtin_amdgcn_wmma_f32_16x16x32_bf16(false, a, false, b,
                                                   (short)0, c, false, false);
}
__device__ inline void async_copy_b128(unsigned lds_byte_addr, const void* gptr) {
    // GLOBAL_LOAD_ASYNC_TO_LDS_B128, GV mode (saddr = off):
    //   copies 16B from global address VADDR to LDS address VDST, tracked by ASYNCcnt
    unsigned long long ga = (unsigned long long)(uintptr_t)gptr;
    asm volatile("global_load_async_to_lds_b128 %0, %1, off"
                 :: "v"(lds_byte_addr), "v"(ga)
                 : "memory");
}
__device__ inline void wait_asynccnt0() {
    asm volatile("s_wait_asynccnt 0x0" ::: "memory");
}

__global__ __launch_bounds__(256)
void sim_gemm_kernel(const __bf16* __restrict__ hi, const __bf16* __restrict__ lo,
                     float* __restrict__ simT) {
    __shared__ __bf16 bsm[2][2][64 * BROW_STRIDE]; // [buf][hi/lo][row*stride+k]

    const int tid  = threadIdx.x;
    const int lane = tid & 31;
    const int wave = tid >> 5;
    const int half = lane >> 4;
    const int mr   = lane & 15;
    const int n0 = blockIdx.x * 64;              // source columns
    const int m0 = blockIdx.y * 128 + wave * 16; // target rows

    // Cooperative B-tile copy: thread tid moves one 16B chunk of hi + one of lo.
    const int crow = tid >> 2;          // 0..63 local B row
    const int coff = (tid & 3) * 8;     // element offset within the 32-K chunk
    const __bf16* gBhi = hi + (size_t)(n0 + crow) * DIM + coff;
    const __bf16* gBlo = lo + (size_t)(n0 + crow) * DIM + coff;
    const unsigned ldshi0 = (unsigned)(uintptr_t)&bsm[0][0][crow * BROW_STRIDE + coff];
    const unsigned ldslo0 = (unsigned)(uintptr_t)&bsm[0][1][crow * BROW_STRIDE + coff];
    const unsigned ldshi1 = (unsigned)(uintptr_t)&bsm[1][0][crow * BROW_STRIDE + coff];
    const unsigned ldslo1 = (unsigned)(uintptr_t)&bsm[1][1][crow * BROW_STRIDE + coff];

    v8f acc[4];
    for (int s = 0; s < 4; ++s)
        for (int e = 0; e < 8; ++e) acc[s][e] = 0.0f;

    // Prologue: stage k-chunk 0 into buffer 0.
    async_copy_b128(ldshi0, gBhi);
    async_copy_b128(ldslo0, gBlo);
    wait_asynccnt0();
    __syncthreads();

    const int arow = N_SRC + m0 + mr;  // target region starts at row 4096
    for (int kk = 0; kk < DIM / 32; ++kk) {
        const int k0  = kk * 32;
        const int buf = kk & 1;
        // Issue async copy of next chunk into the other buffer (overlaps WMMA).
        if (kk + 1 < DIM / 32) {
            if (buf == 0) {
                async_copy_b128(ldshi1, gBhi + k0 + 32);
                async_copy_b128(ldslo1, gBlo + k0 + 32);
            } else {
                async_copy_b128(ldshi0, gBhi + k0 + 32);
                async_copy_b128(ldslo0, gBlo + k0 + 32);
            }
        }

        v16bf ah = load_fragA(hi, arow, k0, half);
        v16bf al = load_fragA(lo, arow, k0, half);
        const __bf16* thi = &bsm[buf][0][0];
        const __bf16* tlo = &bsm[buf][1][0];
#pragma unroll
        for (int s = 0; s < 4; ++s) {
            const int nloc = 16 * s + mr;
            v16bf bh = load_fragB_lds(thi, nloc, half);
            v16bf bl = load_fragB_lds(tlo, nloc, half);
            acc[s] = wmma_bf16(ah, bh, acc[s]);  // hi*hi
            acc[s] = wmma_bf16(ah, bl, acc[s]);  // hi*lo
            acc[s] = wmma_bf16(al, bh, acc[s]);  // lo*hi
        }

        wait_asynccnt0();   // next chunk fully in LDS (per-wave share)
        __syncthreads();    // all waves done with this buffer + all shares visible
    }

    // D layout: VGPR r -> M = r + 8*half, N = mr
#pragma unroll
    for (int s = 0; s < 4; ++s) {
#pragma unroll
        for (int r = 0; r < 8; ++r) {
            int m = m0 + half * 8 + r;
            int n = n0 + 16 * s + mr;
            simT[(size_t)m * N_SRC + n] = 0.5f * (acc[s][r] + 1.0f);
        }
    }
}

// ---------------- top-k helpers --------------------------------------------
__device__ inline bool better(float a, int ia, float b, int ib) {
    return (a > b) || (a == b && ia < ib);
}
template <int R>
__device__ inline void insertTop(float* lv, int* li, float v, int j) {
    if (!better(v, j, lv[R - 1], li[R - 1])) return;
    lv[R - 1] = v; li[R - 1] = j;
    for (int k = R - 1; k > 0; --k) {
        if (better(lv[k], li[k], lv[k - 1], li[k - 1])) {
            float tv = lv[k]; lv[k] = lv[k - 1]; lv[k - 1] = tv;
            int ti = li[k]; li[k] = li[k - 1]; li[k - 1] = ti;
        } else break;
    }
}
template <int R>
__device__ inline void mergeTop(float* av, int* ai, const float* bv, const int* bi) {
    float ov[R]; int oi[R];
    int p = 0, q = 0;
    for (int k = 0; k < R; ++k) {
        bool takeA = (q >= R) || ((p < R) && better(av[p], ai[p], bv[q], bi[q]));
        if (takeA) { ov[k] = av[p]; oi[k] = ai[p]; ++p; }
        else       { ov[k] = bv[q]; oi[k] = bi[q]; ++q; }
    }
    for (int k = 0; k < R; ++k) { av[k] = ov[k]; ai[k] = oi[k]; }
}
template <int R>
__device__ inline void reduceTop(float* rv, int* ri, float* lv, int* li, int t) {
    for (int k = 0; k < R; ++k) { rv[t * R + k] = lv[k]; ri[t * R + k] = li[k]; }
    __syncthreads();
    for (int s = 128; s >= 1; s >>= 1) {
        if (t < s) mergeTop<R>(&rv[t * R], &ri[t * R], &rv[(t + s) * R], &ri[(t + s) * R]);
        __syncthreads();
    }
}

// ---------------- K3+K4: assigned label + ranking score (1 block / target) --
__global__ __launch_bounds__(256)
void rank_kernel(const float* __restrict__ simT, const int* __restrict__ labels,
                 int* __restrict__ assigned, float* __restrict__ score) {
    const int i = blockIdx.x;
    const int t = threadIdx.x;
    __shared__ float srow[N_SRC];
    __shared__ int   slab[N_SRC];
    __shared__ float rv[256 * RK];
    __shared__ int   ri[256 * RK];
    __shared__ int   s_assigned;

    for (int p = 0; p < 16; ++p) {
        int j = p * 256 + t;
        srow[j] = simT[(size_t)i * N_SRC + j];
        slab[j] = labels[j];
    }
    __syncthreads();

    // ---- phase 1: top-5 + majority-vote class ----
    {
        float lv[TOPS]; int li[TOPS];
        for (int k = 0; k < TOPS; ++k) { lv[k] = -3.0e38f; li[k] = 0x7fffffff; }
        for (int p = 0; p < 16; ++p) {
            int j = p * 256 + t;
            insertTop<TOPS>(lv, li, srow[j], j);
        }
        reduceTop<TOPS>(rv, ri, lv, li, t);
        if (t == 0) {
            int lab5[TOPS];
            for (int k = 0; k < TOPS; ++k) lab5[k] = slab[ri[k]];
            int bestc = 0, bestn = -1;
            for (int c = 0; c < NUM_CLS; ++c) {
                int n = 0;
                for (int k = 0; k < TOPS; ++k) n += (lab5[k] == c);
                if (n > bestn) { bestn = n; bestc = c; }
            }
            assigned[i] = bestc;
            s_assigned = bestc;
        }
    }
    __syncthreads();
    const int ai_cls = s_assigned;

    // ---- phase 2: top-10 masked / unmasked sums ----
    float av[RK]; int aidx[RK];
    float bv[RK]; int bidx[RK];
    for (int k = 0; k < RK; ++k) {
        av[k] = -3.0e38f; aidx[k] = 0x7fffffff;
        bv[k] = -3.0e38f; bidx[k] = 0x7fffffff;
    }
    for (int p = 0; p < 16; ++p) {
        int j = p * 256 + t;
        float v = srow[j];
        bool mk = (slab[j] == ai_cls);
        insertTop<RK>(av, aidx, mk ? v : NEG_FILL, j);
        insertTop<RK>(bv, bidx, mk ? NEG_FILL : v, j);
    }
    reduceTop<RK>(rv, ri, av, aidx, t);
    float nln = 0.0f;
    if (t == 0) for (int k = 0; k < RK; ++k) nln += rv[k];
    __syncthreads();
    reduceTop<RK>(rv, ri, bv, bidx, t);
    if (t == 0) {
        float nun = 0.0f;
        for (int k = 0; k < RK; ++k) nun += rv[k];
        score[i] = nln / nun;
    }
}

// ---------------- K5: stable descending rank -> top-2048 indices ------------
__global__ __launch_bounds__(1024)
void select_kernel(const float* __restrict__ score, const int* __restrict__ assigned,
                   int* __restrict__ tind, int* __restrict__ clab,
                   float* __restrict__ accum) {
    const int t = threadIdx.x;
    __shared__ float ss[N_TAR];
    for (int p = 0; p < 4; ++p) ss[p * 1024 + t] = score[p * 1024 + t];
    __syncthreads();
    for (int p = 0; p < 4; ++p) {
        int i = p * 1024 + t;
        float si = ss[i];
        int rank = 0;
        for (int j = 0; j < N_TAR; ++j) {
            float sj = ss[j];
            rank += (sj > si) || (sj == si && j < i);
        }
        if (rank < TOPN) {
            tind[rank] = i;
            clab[rank] = assigned[i];
        }
    }
    if (t == 0) { accum[0] = 0.0f; accum[1] = 0.0f; }
}

// ---------------- K6: online softmax loss over selected columns -------------
__global__ __launch_bounds__(256)
void loss_kernel(const float* __restrict__ simT, const int* __restrict__ labels,
                 const int* __restrict__ tind, const int* __restrict__ clab,
                 float* __restrict__ accum) {
    const int t = threadIdx.x;
    __shared__ int stind[TOPN];
    __shared__ int sclab[TOPN];
    for (int p = 0; p < TOPN / 256; ++p) {
        int j = p * 256 + t;
        stind[j] = tind[j];
        sclab[j] = clab[j];
    }
    __syncthreads();
    const int i = blockIdx.x * 256 + t;
    const int li = labels[i];
    float m = -3.0e38f, l = 0.0f, num = 0.0f;
    bool hasM = false, hasU = false;
    for (int j = 0; j < TOPN; ++j) {
        if (j + 8 < TOPN)
            __builtin_prefetch(&simT[(size_t)stind[j + 8] * N_SRC + i], 0, 0);
        float v = simT[(size_t)stind[j] * N_SRC + i];
        bool mk = (sclab[j] == li);
        hasM |= mk;
        hasU |= !mk;
        float nm = fmaxf(m, v);
        float sc = __expf(m - nm);
        float e  = __expf(v - nm);
        l   = l * sc + e;
        num = num * sc + (mk ? e : 0.0f);
        m = nm;
    }
    if (hasM && hasU) {
        atomicAdd(&accum[0], __logf(num / l));
        atomicAdd(&accum[1], 1.0f);
    }
}

// ---------------- K7: finalize ----------------------------------------------
__global__ void final_kernel(const float* __restrict__ accum, float* __restrict__ out) {
    out[0] = -(accum[0] / accum[1]);
}

// ---------------- host launcher ---------------------------------------------
extern "C" void kernel_launch(void* const* d_in, const int* in_sizes, int n_in,
                              void* d_out, int out_size, void* d_ws, size_t ws_size,
                              hipStream_t stream) {
    const float* x      = (const float*)d_in[0];
    const int*   labels = (const int*)d_in[1];
    float*       out    = (float*)d_out;
    char*        ws     = (char*)d_ws;

    const size_t SZ_BF  = (size_t)(2 * N_SRC) * DIM * sizeof(__bf16); // 16 MB each
    const size_t SZ_SIM = (size_t)N_TAR * N_SRC * sizeof(float);      // 64 MB

    __bf16* hi       = (__bf16*)(ws);
    __bf16* lo       = (__bf16*)(ws + SZ_BF);
    float*  simT     = (float*)(ws + 2 * SZ_BF);
    char*   small    = ws + 2 * SZ_BF + SZ_SIM;
    int*    assigned = (int*)(small);
    float*  score    = (float*)(small + 16384);
    int*    tind     = (int*)(small + 32768);
    int*    clab     = (int*)(small + 32768 + 8192);
    float*  accum    = (float*)(small + 32768 + 16384);

    normalize_split_kernel<<<2 * N_SRC, 256, 0, stream>>>(x, hi, lo);

    dim3 gemm_grid(N_SRC / 64, N_TAR / 128);
    sim_gemm_kernel<<<gemm_grid, 256, 0, stream>>>(hi, lo, simT);

    rank_kernel<<<N_TAR, 256, 0, stream>>>(simT, labels, assigned, score);

    select_kernel<<<1, 1024, 0, stream>>>(score, assigned, tind, clab, accum);

    loss_kernel<<<N_SRC / 256, 256, 0, stream>>>(simT, labels, tind, clab, accum);

    final_kernel<<<1, 1, 0, stream>>>(accum, out);
}
